// GeneratorCell_16389595202100
// MI455X (gfx1250) — matrix operand order repaired and checked by reading, not compiled, not run on hardware
//
#include <hip/hip_runtime.h>
#include <hip/hip_fp16.h>

typedef __attribute__((ext_vector_type(16))) _Float16 v16h;
typedef __attribute__((ext_vector_type(8)))  _Float16 v8h;
typedef __attribute__((ext_vector_type(8)))  float    v8f;

#define BB 2048
#define NN 1024
#define VV 100000

// ---------------------------------------------------------------------------
// Prep: A1 = f16(relu(cat(conv1d(st1), ht1)))  [B, 2N]
// ---------------------------------------------------------------------------
__global__ void prep_mt1(const float* __restrict__ st1, const float* __restrict__ ht1,
                         const float* __restrict__ conv_w, const float* __restrict__ conv_b,
                         _Float16* __restrict__ A1) {
    int idx = blockIdx.x * blockDim.x + threadIdx.x;   // over B*2N
    if (idx >= BB * 2 * NN) return;
    int b = idx / (2 * NN);
    int i = idx % (2 * NN);
    float v;
    if (i < NN) {
        float acc = conv_b[0];
#pragma unroll
        for (int j = 0; j < 5; ++j) {
            int p = i - 2 + j;
            float x = (p >= 0 && p < NN) ? st1[(size_t)b * NN + p] : 0.f;
            acc += conv_w[j] * x;
        }
        v = acc;
    } else {
        v = ht1[(size_t)b * NN + (i - NN)];
    }
    A1[idx] = (_Float16)fmaxf(v, 0.f);
}

// ---------------------------------------------------------------------------
// Prep: A2 = f16(cat(zt, da))  [B, N]
// ---------------------------------------------------------------------------
__global__ void prep_xt(const float* __restrict__ zt, const int* __restrict__ da,
                        _Float16* __restrict__ A2) {
    int idx = blockIdx.x * blockDim.x + threadIdx.x;   // over B*N
    if (idx >= BB * NN) return;
    int b = idx / NN;
    int i = idx % NN;
    float v = (i < NN - 1) ? zt[(size_t)b * (NN - 1) + i] : (float)da[b];
    A2[idx] = (_Float16)v;
}

// ---------------------------------------------------------------------------
// Pack W [Ncols, K] (row-major, f32) into WMMA B-fragment order, f16.
// Tile (kblock, ntile) of 32x16; lane L holds col n = L%16,
// K = (t/8)*16 + (L/16)*8 + t%8, t = 0..15 (32 bytes per lane).
// ---------------------------------------------------------------------------
__global__ void pack_w(const float* __restrict__ W, _Float16* __restrict__ P,
                       int Ncols, int K) {
    int e = blockIdx.x * blockDim.x + threadIdx.x;
    if (e >= Ncols * K) return;
    int tIdx = e >> 9;            // /512 halfs per 32x16 tile
    int r    = e & 511;
    int L    = r >> 4;
    int t    = r & 15;
    int ntiles = Ncols >> 4;
    int kblock = tIdx / ntiles;
    int ntile  = tIdx % ntiles;
    int k = kblock * 32 + (t >> 3) * 16 + (L >> 4) * 8 + (t & 7);
    int n = ntile * 16 + (L & 15);
    P[e] = (_Float16)W[(size_t)n * K + k];
}

// ---------------------------------------------------------------------------
// WMMA GEMM: C[M,Ncols] = A[M,K](f16) * Bpacked + bias, f32 accumulate.
// 8 waves/block, 128x128 block tile, 64x32 per wave (4x2 WMMA tiles).
// WRITE_HALF additionally emits an f16 copy of C (for chained GEMMs).
// ---------------------------------------------------------------------------
template <bool WRITE_HALF>
__global__ void __launch_bounds__(256)
gemm_wmma(const _Float16* __restrict__ A, const _Float16* __restrict__ Bp,
          float* __restrict__ C, _Float16* __restrict__ Ch,
          const float* __restrict__ bias, int M, int Ncols, int K) {
    const int lane = threadIdx.x & 31;
    const int wid  = threadIdx.x >> 5;
    const int wm   = wid & 1;           // 2 waves in M
    const int wn   = wid >> 1;          // 4 waves in N
    const int m0   = blockIdx.y * 128 + wm * 64;
    const int n0   = blockIdx.x * 128 + wn * 32;
    const int half = lane >> 4;
    const int l16  = lane & 15;
    const int ntiles = Ncols >> 4;

    v8f acc[4][2] = {};

    for (int kb = 0; kb < K; kb += 32) {
        // A fragments: ISA 16-bit A layout; per lane two contiguous b128 loads
        v16h a[4];
#pragma unroll
        for (int i = 0; i < 4; ++i) {
            const _Float16* pa = A + (size_t)(m0 + i * 16 + l16) * K + kb + half * 8;
            v8h lo = *(const v8h*)pa;          // K group 0..15 slice
            v8h hi = *(const v8h*)(pa + 16);   // K group 16..31 slice
#pragma unroll
            for (int t = 0; t < 8; ++t) { a[i][t] = lo[t]; a[i][8 + t] = hi[t]; }
            // WGP-scope prefetch of next K-block (locality 3 -> fill L0/WGP$)
            __builtin_prefetch((const void*)(pa + 32), 0, 3);
        }
        // B fragments: pre-packed, 32 contiguous bytes per lane
        v16h bfr[2];
#pragma unroll
        for (int j = 0; j < 2; ++j) {
            size_t tIdx = (size_t)(kb >> 5) * ntiles + ((n0 >> 4) + j);
            bfr[j] = *(const v16h*)(Bp + tIdx * 512 + lane * 16);
            __builtin_prefetch((const void*)(Bp + (tIdx + ntiles) * 512 + lane * 16), 0, 3);
        }
#pragma unroll
        for (int i = 0; i < 4; ++i)
#pragma unroll
            for (int j = 0; j < 2; ++j)
                acc[i][j] = __builtin_amdgcn_wmma_f32_16x16x32_f16(
                    false, a[i], false, bfr[j], (short)0, acc[i][j], false, false);
    }

    // Epilogue: D layout row = v + half*8, col = l16 (branch-free)
#pragma unroll
    for (int i = 0; i < 4; ++i) {
#pragma unroll
        for (int j = 0; j < 2; ++j) {
            int col = n0 + j * 16 + l16;
            float bv = bias[col];
#pragma unroll
            for (int v = 0; v < 8; ++v) {
                int row = m0 + i * 16 + v + half * 8;
                float d = acc[i][j][v] + bv;
                C[(size_t)row * Ncols + col] = d;
                if constexpr (WRITE_HALF) {
                    Ch[(size_t)row * Ncols + col] = (_Float16)d;
                }
            }
        }
    }
}

// ---------------------------------------------------------------------------
// Per-row GRU gate math + reductions + gather + shift epilogue.
// One 256-thread block per batch row.
// ---------------------------------------------------------------------------
__global__ void __launch_bounds__(256)
gru_epilogue(const float* __restrict__ gi, const float* __restrict__ gh,
             const float* __restrict__ Ht1,
             const float* __restrict__ st1, const int* __restrict__ da_t,
             const float* __restrict__ W_ht, const float* __restrict__ b_ht,
             const float* __restrict__ emb,
             const float* __restrict__ W_ct, const float* __restrict__ b_ct,
             float* __restrict__ out_st, float* __restrict__ out_ht) {
    const int b   = blockIdx.x;
    const int tid = threadIdx.x;
    __shared__ float red1[256];
    __shared__ float red2[256];
    __shared__ float bc[2];

    float s1 = 0.f, s2 = 0.f;
#pragma unroll
    for (int q = 0; q < 4; ++q) {
        int i = tid + q * 256;
        size_t o = (size_t)b * 3 * NN + i;
        float gir = gi[o], giz = gi[o + NN], gin = gi[o + 2 * NN];
        float ghr = gh[o], ghz = gh[o + NN], ghn = gh[o + 2 * NN];
        float r  = 1.f / (1.f + expf(-(gir + ghr)));
        float z  = 1.f / (1.f + expf(-(giz + ghz)));
        float nv = tanhf(gin + r * ghn);
        float h0 = Ht1[(size_t)b * NN + i];
        float h  = (1.f - z) * nv + z * h0;
        out_ht[(size_t)b * NN + i] = h;
        s1 += h * W_ht[i];
        s2 += fmaxf(h, 0.f) * W_ct[i];
    }
    red1[tid] = s1;
    red2[tid] = s2;
    __syncthreads();
    for (int s = 128; s > 0; s >>= 1) {
        if (tid < s) { red1[tid] += red1[tid + s]; red2[tid] += red2[tid + s]; }
        __syncthreads();
    }
    if (tid == 0) {
        float logits = red1[0] + b_ht[0];
        float sg = 1.f / (1.f + expf(-logits));
        int it = (int)floorf((float)(VV - 1) * sg);
        it = it < 0 ? 0 : (it > VV - 1 ? VV - 1 : it);
        float daf  = (float)da_t[b];
        float cpre = red2[0] + fmaxf(daf, 0.f) * W_ct[NN] + b_ct[0];
        bc[0] = __int_as_float(it);
        bc[1] = 1.f / (1.f + expf(-cpre));
    }
    __syncthreads();
    const int   it = __float_as_int(bc[0]);
    const float ct = bc[1];
    const int   da = da_t[b];
#pragma unroll
    for (int q = 0; q < 4; ++q) {
        int i   = tid + q * 256;
        int idx = i + da;
        float sh   = (idx < NN) ? st1[(size_t)b * NN + idx] : 0.f;
        float frag = sh * ct;
        out_st[(size_t)b * NN + i] = frag + emb[(size_t)it * NN + i];
    }
}

// ---------------------------------------------------------------------------
extern "C" void kernel_launch(void* const* d_in, const int* in_sizes, int n_in,
                              void* d_out, int out_size, void* d_ws, size_t ws_size,
                              hipStream_t stream) {
    (void)in_sizes; (void)n_in; (void)out_size; (void)ws_size;

    const float* st1    = (const float*)d_in[0];
    const float* ht1    = (const float*)d_in[1];
    const float* zt     = (const float*)d_in[2];
    const int*   da_t   = (const int*)d_in[3];
    const float* conv_w = (const float*)d_in[4];
    const float* conv_b = (const float*)d_in[5];
    const float* W_mt1  = (const float*)d_in[6];
    const float* b_mt1  = (const float*)d_in[7];
    const float* W_ih   = (const float*)d_in[8];
    const float* b_ih   = (const float*)d_in[9];
    const float* W_hh   = (const float*)d_in[10];
    const float* b_hh   = (const float*)d_in[11];
    const float* W_ht   = (const float*)d_in[12];
    const float* b_ht   = (const float*)d_in[13];
    const float* emb    = (const float*)d_in[14];
    const float* W_ct   = (const float*)d_in[15];
    const float* b_ct   = (const float*)d_in[16];

    char* ws = (char*)d_ws;
    size_t off = 0;
    auto carve = [&](size_t bytes) -> void* {
        void* p = ws + off;
        off += (bytes + 255) & ~(size_t)255;
        return p;
    };

    _Float16* A1    = (_Float16*)carve((size_t)BB * 2 * NN * 2);   // relu(Mt1) f16
    _Float16* A2    = (_Float16*)carve((size_t)BB * NN * 2);       // xt f16
    _Float16* pW1   = (_Float16*)carve((size_t)NN * 2 * NN * 2);   // packed W_mt1
    _Float16* pWih  = (_Float16*)carve((size_t)3 * NN * NN * 2);   // packed W_ih
    _Float16* pWhh  = (_Float16*)carve((size_t)3 * NN * NN * 2);   // packed W_hh
    float*    Ht1f  = (float*)   carve((size_t)BB * NN * 4);       // Ht1 f32
    _Float16* Ht1h  = (_Float16*)carve((size_t)BB * NN * 2);       // Ht1 f16
    float*    gi    = (float*)   carve((size_t)BB * 3 * NN * 4);
    float*    gh    = (float*)   carve((size_t)BB * 3 * NN * 4);

    dim3 blk(256);
    prep_mt1<<<(BB * 2 * NN + 255) / 256, blk, 0, stream>>>(st1, ht1, conv_w, conv_b, A1);
    prep_xt <<<(BB * NN + 255) / 256,     blk, 0, stream>>>(zt, da_t, A2);
    pack_w  <<<(NN * 2 * NN + 255) / 256, blk, 0, stream>>>(W_mt1, pW1, NN, 2 * NN);
    pack_w  <<<(3 * NN * NN + 255) / 256, blk, 0, stream>>>(W_ih, pWih, 3 * NN, NN);
    pack_w  <<<(3 * NN * NN + 255) / 256, blk, 0, stream>>>(W_hh, pWhh, 3 * NN, NN);

    // GEMM1: Ht1 = relu(Mt1) @ W_mt1.T + b_mt1   (also f16 copy for GEMM3)
    gemm_wmma<true><<<dim3(NN / 128, BB / 128), blk, 0, stream>>>(
        A1, pW1, Ht1f, Ht1h, b_mt1, BB, NN, 2 * NN);
    // GEMM2: gi = xt @ W_ih.T + b_ih
    gemm_wmma<false><<<dim3(3 * NN / 128, BB / 128), blk, 0, stream>>>(
        A2, pWih, gi, (_Float16*)nullptr, b_ih, BB, 3 * NN, NN);
    // GEMM3: gh = Ht1 @ W_hh.T + b_hh
    gemm_wmma<false><<<dim3(3 * NN / 128, BB / 128), blk, 0, stream>>>(
        Ht1h, pWhh, gh, (_Float16*)nullptr, b_hh, BB, 3 * NN, NN);

    float* out_st = (float*)d_out;
    float* out_ht = out_st + (size_t)BB * NN;
    gru_epilogue<<<BB, blk, 0, stream>>>(gi, gh, Ht1f, st1, da_t, W_ht, b_ht,
                                         emb, W_ct, b_ct, out_st, out_ht);
}
